// TK_v6_relative_42674795053886
// MI455X (gfx1250) — matrix operand order repaired and checked by reading, not compiled, not run on hardware
//
#include <hip/hip_runtime.h>
#include <cstdint>
#include <cstddef>

typedef __attribute__((ext_vector_type(16))) _Float16 v16h;
typedef __attribute__((ext_vector_type(8)))  _Float16 v8h;
typedef __attribute__((ext_vector_type(8)))  float    v8f;

#define D_MODEL 256
#define NHEAD   8
#define NKER    11

__constant__ float c_mu[NKER]   = {1.0f,0.9f,0.7f,0.5f,0.3f,0.1f,-0.1f,-0.3f,-0.5f,-0.7f,-0.9f};
__constant__ float c_i2s2[NKER] = {500000.0f,50.f,50.f,50.f,50.f,50.f,50.f,50.f,50.f,50.f,50.f};

static __device__ __forceinline__ v16h cat16(v8h lo, v8h hi) {
    return __builtin_shufflevector(lo, hi, 0,1,2,3,4,5,6,7,8,9,10,11,12,13,14,15);
}

// ---------------------------------------------------------------- utilities
__global__ void zero_f32(float* p, int n) {
    int i = blockIdx.x * 256 + threadIdx.x;
    if (i < n) p[i] = 0.f;
}

// transpose + convert: d[N,K] = (f16) s[K,N]
__global__ void cvt_tr_f16(const float* __restrict__ s, _Float16* __restrict__ d,
                           int K, int N) {
    int i = blockIdx.x * 256 + threadIdx.x;
    if (i < N * K) {
        int n = i / K, k = i % K;
        d[i] = (_Float16)s[(size_t)k * N + n];
    }
}

// semb = emb*mask ; x = semb ; x16 = f16(semb).  grid = rows, block = 256
__global__ __launch_bounds__(256) void embed_mask_k(
    const float* __restrict__ emb, const float* __restrict__ mask,
    float* __restrict__ semb, float* __restrict__ x, _Float16* __restrict__ x16)
{
    size_t i = (size_t)blockIdx.x * 256 + threadIdx.x;
    size_t r = i >> 8;
    float m = mask[r];
    float v = emb[i] * m;
    semb[i] = v; x[i] = v; x16[i] = (_Float16)v;
}

// ---------------------------------------------------------------- WMMA GEMM
// C[M,N] = A16[M,K] @ Bt16[N,K]^T + bias.  LDS-free: fragments are contiguous
// global vector loads.  Block tile 64x128, 8 waves, each wave a 32x32 tile:
// per K-step 2 A-frags + 2 B-frags feed 4 WMMAs (register-level reuse).
// Optional outputs: Cf (f32), Ch (f16), Cvt (f16 transposed per (b,head) for V).
__global__ __launch_bounds__(256) void gemm16(
    const _Float16* __restrict__ A, const _Float16* __restrict__ Bt,
    const float* __restrict__ bias, float* __restrict__ Cf,
    _Float16* __restrict__ Ch, _Float16* __restrict__ Cvt,
    int M, int N, int K, int relu, int Ls)
{
    const int tid = threadIdx.x;
    const int bm = blockIdx.y * 64, bn = blockIdx.x * 128;
    const int w = tid >> 5, lane = tid & 31;
    const int wm = (w & 1) * 32, wn = (w >> 1) * 32;
    const int n0 = lane & 15, hi = lane >> 4;

    const _Float16* ar0 = A  + (size_t)(bm + wm + n0) * K + hi * 16;
    const _Float16* ar1 = ar0 + (size_t)16 * K;
    const _Float16* br0 = Bt + (size_t)(bn + wn + n0) * K + hi * 16;
    const _Float16* br1 = br0 + (size_t)16 * K;

    v8f cc[2][2];
    #pragma unroll
    for (int i = 0; i < 2; ++i)
        #pragma unroll
        for (int j = 0; j < 2; ++j) cc[i][j] = (v8f){};

    #pragma unroll 2
    for (int k0 = 0; k0 < K; k0 += 32) {
        v16h a0 = cat16(*(const v8h*)(ar0 + k0), *(const v8h*)(ar0 + k0 + 8));
        v16h a1 = cat16(*(const v8h*)(ar1 + k0), *(const v8h*)(ar1 + k0 + 8));
        v16h b0 = *(const v16h*)(br0 + k0);
        v16h b1 = *(const v16h*)(br1 + k0);
        cc[0][0] = __builtin_amdgcn_wmma_f32_16x16x32_f16(false, a0, false, b0,
                                                          (short)0, cc[0][0], false, false);
        cc[0][1] = __builtin_amdgcn_wmma_f32_16x16x32_f16(false, a0, false, b1,
                                                          (short)0, cc[0][1], false, false);
        cc[1][0] = __builtin_amdgcn_wmma_f32_16x16x32_f16(false, a1, false, b0,
                                                          (short)0, cc[1][0], false, false);
        cc[1][1] = __builtin_amdgcn_wmma_f32_16x16x32_f16(false, a1, false, b1,
                                                          (short)0, cc[1][1], false, false);
    }

    #pragma unroll
    for (int nt = 0; nt < 2; ++nt) {
        const int col = bn + wn + nt * 16 + n0;
        const float bv = bias ? bias[col] : 0.f;
        #pragma unroll
        for (int mt = 0; mt < 2; ++mt) {
            #pragma unroll
            for (int r = 0; r < 8; ++r) {
                const int row = bm + wm + mt * 16 + r + hi * 8;
                float v = cc[mt][nt][r] + bv;
                if (relu) v = fmaxf(v, 0.f);
                if (Cf) Cf[(size_t)row * N + col] = v;
                if (Ch) Ch[(size_t)row * N + col] = (_Float16)v;
                if (Cvt) {   // vt[b][h][dh][l] for attention V
                    const int bb = row >> Ls, ll = row & ((1 << Ls) - 1);
                    const int hh = col >> 5,  dh = col & 31;
                    Cvt[((((size_t)bb * NHEAD + hh) * 32 + dh) << Ls) | ll] = (_Float16)v;
                }
            }
        }
    }
}

// ---------------------------------------------------------------- attention
// One block = (b, head, 16-query tile). 4 waves split the 32-key chunks.
// Two-pass softmax (rowmax, then exp/sum + P@V), WMMA for S and P@V.
// V is consumed pre-transposed: vt[b][h][dh][key].
__global__ __launch_bounds__(128) void attn_k(
    const _Float16* __restrict__ q16, const _Float16* __restrict__ k16,
    const _Float16* __restrict__ vt16, const float* __restrict__ mask,
    _Float16* __restrict__ out16, int L)
{
    const int numQT = L >> 4;
    const int qt = blockIdx.x % numQT;
    const int bh = blockIdx.x / numQT;
    const int h  = bh % NHEAD, b = bh / NHEAD;
    const int tid = threadIdx.x, w = tid >> 5, lane = tid & 31;
    const int n0 = lane & 15, hi = lane >> 4;
    const int row0 = b * L + qt * 16;
    const int hoff = h * 32;
    const float scale = 0.17677669529663687f;

    __shared__ float    sh_m[4][16];
    __shared__ float    sh_l[4][16];
    __shared__ float    sh_g[16];
    __shared__ float    sh_gl[16];
    __shared__ float    sh_o[4][16][32];
    __shared__ _Float16 sh_p[4][16][32];

    // Q A-fragment (16x32 f16), identical per wave
    v16h aq;
    {
        const _Float16* qp = q16 + (size_t)(row0 + n0) * D_MODEL + hoff + hi * 16;
        aq = cat16(*(const v8h*)qp, *(const v8h*)(qp + 8));
    }
    const _Float16* vtb = vt16 + (((size_t)b * NHEAD + h) * 32) * L;

    float rm[8], rl[8];
    #pragma unroll
    for (int r = 0; r < 8; ++r) { rm[r] = -3.0e38f; rl[r] = 0.f; }
    v8f o0 = {}, o1 = {};
    const int nChunk = L >> 5;

    // ---- pass 1: row max
    for (int ch = w; ch < nChunk; ch += 4) {
        const int kb = ch * 32;
        #pragma unroll
        for (int t = 0; t < 2; ++t) {
            const int key = kb + t * 16 + n0;
            v16h bk = *(const v16h*)(k16 + (size_t)(b * L + key) * D_MODEL + hoff + hi * 16);
            v8f s = {};
            s = __builtin_amdgcn_wmma_f32_16x16x32_f16(false, aq, false, bk,
                                                       (short)0, s, false, false);
            const float mval = mask[b * L + key];
            #pragma unroll
            for (int r = 0; r < 8; ++r) {
                float sv = (mval > 0.f) ? s[r] * scale : -1.0e9f;
                rm[r] = fmaxf(rm[r], sv);
            }
        }
    }
    #pragma unroll
    for (int r = 0; r < 8; ++r) {
        float v = rm[r];
        for (int off = 1; off < 16; off <<= 1) v = fmaxf(v, __shfl_xor(v, off, 16));
        rm[r] = v;
    }
    if (n0 == 0) {
        #pragma unroll
        for (int r = 0; r < 8; ++r) sh_m[w][r + hi * 8] = rm[r];
    }
    __syncthreads();
    if (tid < 16)
        sh_g[tid] = fmaxf(fmaxf(sh_m[0][tid], sh_m[1][tid]),
                          fmaxf(sh_m[2][tid], sh_m[3][tid]));
    __syncthreads();

    // ---- pass 2: exp, sums, P@V
    for (int ch = w; ch < nChunk; ch += 4) {
        const int kb = ch * 32;
        #pragma unroll
        for (int t = 0; t < 2; ++t) {
            const int key = kb + t * 16 + n0;
            v16h bk = *(const v16h*)(k16 + (size_t)(b * L + key) * D_MODEL + hoff + hi * 16);
            v8f s = {};
            s = __builtin_amdgcn_wmma_f32_16x16x32_f16(false, aq, false, bk,
                                                       (short)0, s, false, false);
            const float mval = mask[b * L + key];
            #pragma unroll
            for (int r = 0; r < 8; ++r) {
                const int rowr = r + hi * 8;
                float sv = (mval > 0.f) ? s[r] * scale : -1.0e9f;
                float p  = __expf(sv - sh_g[rowr]);
                rl[r] += p;
                sh_p[w][rowr][t * 16 + n0] = (_Float16)p;
            }
        }
        asm volatile("s_wait_dscnt 0" ::: "memory");
        v16h ap;
        {
            const v8h* pp = (const v8h*)(&sh_p[w][n0][hi * 16]);
            ap = cat16(pp[0], pp[1]);      // +0 and +8 halves
        }
        // V fragments: contiguous along key thanks to vt layout
        v16h bv0 = *(const v16h*)(vtb + (size_t)n0 * L + kb + hi * 16);
        v16h bv1 = *(const v16h*)(vtb + (size_t)(16 + n0) * L + kb + hi * 16);
        o0 = __builtin_amdgcn_wmma_f32_16x16x32_f16(false, ap, false, bv0,
                                                    (short)0, o0, false, false);
        o1 = __builtin_amdgcn_wmma_f32_16x16x32_f16(false, ap, false, bv1,
                                                    (short)0, o1, false, false);
    }

    #pragma unroll
    for (int r = 0; r < 8; ++r) {
        float v = rl[r];
        for (int off = 1; off < 16; off <<= 1) v += __shfl_xor(v, off, 16);
        rl[r] = v;
    }
    if (n0 == 0) {
        #pragma unroll
        for (int r = 0; r < 8; ++r) sh_l[w][r + hi * 8] = rl[r];
    }
    #pragma unroll
    for (int r = 0; r < 8; ++r) {
        sh_o[w][r + hi * 8][n0]      = o0[r];
        sh_o[w][r + hi * 8][16 + n0] = o1[r];
    }
    __syncthreads();
    if (tid < 16)
        sh_gl[tid] = sh_l[0][tid] + sh_l[1][tid] + sh_l[2][tid] + sh_l[3][tid];
    __syncthreads();
    for (int idx = tid; idx < 512; idx += 128) {
        const int rr = idx >> 5, cc = idx & 31;
        float val = sh_o[0][rr][cc] + sh_o[1][rr][cc] + sh_o[2][rr][cc] + sh_o[3][rr][cc];
        out16[(size_t)(row0 + rr) * D_MODEL + hoff + cc] = (_Float16)(val / sh_gl[rr]);
    }
}

// ---------------------------------------------------------------- residual+LN
__global__ __launch_bounds__(256) void add_ln_k(
    float* __restrict__ x, const float* __restrict__ y,
    const float* __restrict__ g, const float* __restrict__ bb,
    _Float16* __restrict__ x16)
{
    __shared__ float red[256];
    const int row = blockIdx.x, t = threadIdx.x;
    const size_t idx = (size_t)row * 256 + t;
    float v = x[idx] + y[idx];
    red[t] = v; __syncthreads();
    for (int s = 128; s > 0; s >>= 1) { if (t < s) red[t] += red[t + s]; __syncthreads(); }
    const float mu = red[0] * (1.f / 256.f);
    __syncthreads();
    const float d = v - mu;
    red[t] = d * d; __syncthreads();
    for (int s = 128; s > 0; s >>= 1) { if (t < s) red[t] += red[t + s]; __syncthreads(); }
    const float var = red[0] * (1.f / 256.f);
    const float nv = d * rsqrtf(var + 1e-5f) * g[t] + bb[t];
    x[idx] = nv; x16[idx] = (_Float16)nv;
}

// ---------------------------------------------------------------- mixer + L2 norm
__global__ __launch_bounds__(256) void mix_norm_k(
    const float* __restrict__ semb, const float* __restrict__ ctx,
    const float* __restrict__ mask, const float* __restrict__ mixer,
    _Float16* __restrict__ n16)
{
    __shared__ float red[256];
    const int row = blockIdx.x, t = threadIdx.x;
    const float m = mixer[0], msk = mask[row];
    const size_t idx = (size_t)row * 256 + t;
    const float v = (m * semb[idx] + (1.f - m) * ctx[idx]) * msk;
    red[t] = v * v; __syncthreads();
    for (int s = 128; s > 0; s >>= 1) { if (t < s) red[t] += red[t + s]; __syncthreads(); }
    const float norm = sqrtf(red[0]) + 1e-13f;
    n16[idx] = (_Float16)(v / norm);
}

// ---------------------------------------------------------------- positional bias multiplier
__global__ __launch_bounds__(256) void pos_mult_k(
    const float* __restrict__ dmask, const float* __restrict__ pb,
    const float* __restrict__ pba, float* __restrict__ mult)
{
    __shared__ float red[256];
    __shared__ int s_w;
    const int b = blockIdx.x, t = threadIdx.x;
    float acc = 0.f;
    for (int d = t; d < 1024; d += 256) acc += dmask[b * 1024 + d];
    red[t] = acc; __syncthreads();
    for (int s = 128; s > 0; s >>= 1) { if (t < s) red[t] += red[t + s]; __syncthreads(); }
    if (t == 0) s_w = (int)rintf(red[0] * 0.1f);
    __syncthreads();
    const int wbin = s_w;
    for (int d = t; d < 1024; d += 256) {
        const float mk = dmask[b * 1024 + d];
        int cum = (wbin == 0) ? 1 : (d / wbin + 1);
        if (cum > 10) cum = 10;
        const int pidx = (mk > 0.f) ? cum : 0;
        const int aidx = (mk > 0.f) ? (d / 100 + 1) : 0;
        mult[b * 1024 + d] = pb[pidx] * pba[aidx];
    }
}

// ---------------------------------------------------------------- kernel pooling / scoring
__global__ __launch_bounds__(256) void score_k(
    const _Float16* __restrict__ qn16, const _Float16* __restrict__ dn16,
    const float* __restrict__ qmask, const float* __restrict__ mult,
    float* __restrict__ per_kernel)
{
    __shared__ float qrow[256];
    __shared__ float sacc[16];
    const int bq = blockIdx.x;          // b*32 + q
    const int b  = bq >> 5;
    const int t  = threadIdx.x;
    qrow[t] = (float)qn16[(size_t)bq * 256 + t];
    if (t < 16) sacc[t] = 0.f;
    __syncthreads();

    float acc[NKER];
    #pragma unroll
    for (int nk = 0; nk < NKER; ++nk) acc[nk] = 0.f;

    for (int d = t; d < 1024; d += 256) {
        const float mlt = mult[b * 1024 + d];
        if (mlt == 0.f) continue;
        const v16h* dp = (const v16h*)(dn16 + ((size_t)b * 1024 + d) * 256);
        float c = 0.f;
        #pragma unroll 4
        for (int g = 0; g < 16; ++g) {
            v16h dv = dp[g];
            #pragma unroll
            for (int i = 0; i < 16; ++i) c += qrow[g * 16 + i] * (float)dv[i];
        }
        #pragma unroll
        for (int nk = 0; nk < NKER; ++nk) {
            const float dm = c - c_mu[nk];
            acc[nk] += __expf(-dm * dm * c_i2s2[nk]) * mlt;
        }
    }
    #pragma unroll
    for (int nk = 0; nk < NKER; ++nk) atomicAdd(&sacc[nk], acc[nk]);
    __syncthreads();
    if (t < NKER) {
        const float lv = logf(fmaxf(sacc[t], 1e-10f)) * qmask[bq];
        atomicAdd(&per_kernel[b * NKER + t], lv);
    }
}

__global__ void final_k(const float* __restrict__ per_kernel,
                        const float* __restrict__ dw, float* __restrict__ out)
{
    const int b = threadIdx.x;
    if (b < 32) {
        float s = 0.f;
        for (int k = 0; k < NKER; ++k) s += per_kernel[b * NKER + k] * dw[k];
        out[b] = s;
    }
}

// ================================================================ host
extern "C" void kernel_launch(void* const* d_in, const int* in_sizes, int n_in,
                              void* d_out, int out_size, void* d_ws, size_t ws_size,
                              hipStream_t stream)
{
    (void)n_in; (void)out_size; (void)ws_size;
    const float* q_emb  = (const float*)d_in[0];
    const float* d_emb  = (const float*)d_in[1];
    const float* q_mask = (const float*)d_in[2];
    const float* d_mask = (const float*)d_in[3];

    // Per-layer param pointers in canonical order:
    // Wq bq Wk bk Wv bv Wo bo W1 b1 W2 b2 ln1_g ln1_b ln2_g ln2_b
    const float* P[2][16];
    const float *mixer, *pb, *pba, *dw;
    if (in_sizes[4] == 65536) {
        // insertion-order flattening
        for (int l = 0; l < 2; ++l)
            for (int j = 0; j < 16; ++j) P[l][j] = (const float*)d_in[4 + l * 16 + j];
        mixer = (const float*)d_in[36]; pb = (const float*)d_in[37];
        pba = (const float*)d_in[38];   dw = (const float*)d_in[39];
    } else {
        // jax tree_leaves (alphabetical) flattening:
        // top: dense_w, layers[0..1], mixer, position_bias, position_bias_absolute
        // layer: W1 W2 Wk Wo Wq Wv b1 b2 bk bo bq bv ln1_b ln1_g ln2_b ln2_g
        dw = (const float*)d_in[4];
        static const int amap[16] = { /*Wq*/4,/*bq*/10,/*Wk*/2,/*bk*/8,/*Wv*/5,/*bv*/11,
                                      /*Wo*/3,/*bo*/9,/*W1*/0,/*b1*/6,/*W2*/1,/*b2*/7,
                                      /*ln1_g*/13,/*ln1_b*/12,/*ln2_g*/15,/*ln2_b*/14 };
        for (int l = 0; l < 2; ++l)
            for (int j = 0; j < 16; ++j) P[l][j] = (const float*)d_in[5 + l * 16 + amap[j]];
        mixer = (const float*)d_in[37]; pb = (const float*)d_in[38];
        pba = (const float*)d_in[39];
    }

    char* ws = (char*)d_ws;
    size_t off = 0;
    auto alloc = [&](size_t bytes) -> void* {
        void* p = (void*)(ws + off);
        off += (bytes + 255) & ~(size_t)255;
        return p;
    };

    const int B = 32, QL = 32, DL = 1024, Dm = D_MODEL;
    const int MQ = B * QL, MD = B * DL;

    // f16 transposed weights [N,K]: Wq Wk Wv Wo W1 W2 per layer
    static const int wK[6] = {256, 256, 256, 256, 256, 1024};
    static const int wN[6] = {256, 256, 256, 256, 1024, 256};
    static const int wj[6] = {0, 2, 4, 6, 8, 10};
    _Float16* wt16[2][6];
    for (int l = 0; l < 2; ++l)
        for (int j = 0; j < 6; ++j) wt16[l][j] = (_Float16*)alloc((size_t)wK[j] * wN[j] * 2);

    float*    sembf = (float*)   alloc((size_t)MD * Dm * 4);
    float*    xf    = (float*)   alloc((size_t)MD * Dm * 4);
    _Float16* x16   = (_Float16*)alloc((size_t)MD * Dm * 2);
    _Float16* q16   = (_Float16*)alloc((size_t)MD * Dm * 2);
    _Float16* k16   = (_Float16*)alloc((size_t)MD * Dm * 2);
    _Float16* vt16  = (_Float16*)alloc((size_t)MD * Dm * 2);
    _Float16* a16   = (_Float16*)alloc((size_t)MD * Dm * 2);
    float*    tmpf  = (float*)   alloc((size_t)MD * Dm * 4);
    _Float16* h16   = (_Float16*)alloc((size_t)MD * 1024 * 2);
    _Float16* qn16  = (_Float16*)alloc((size_t)MQ * Dm * 2);
    _Float16* dn16  = (_Float16*)alloc((size_t)MD * Dm * 2);
    float*    multb = (float*)   alloc((size_t)B * DL * 4);
    float*    pk    = (float*)   alloc((size_t)B * NKER * 4);

    // convert+transpose weights fp32[K,N] -> f16[N,K]
    for (int l = 0; l < 2; ++l)
        for (int j = 0; j < 6; ++j) {
            int n = wK[j] * wN[j];
            cvt_tr_f16<<<(n + 255) / 256, 256, 0, stream>>>(P[l][wj[j]], wt16[l][j],
                                                            wK[j], wN[j]);
        }

    auto run_encoder = [&](const float* emb, const float* msk, int L, int Ls,
                           _Float16* nout) {
        const int M = B * L;
        embed_mask_k<<<M, 256, 0, stream>>>(emb, msk, sembf, xf, x16);
        for (int l = 0; l < 2; ++l) {
            gemm16<<<dim3(Dm / 128, M / 64), 256, 0, stream>>>(
                x16, wt16[l][0], P[l][1], nullptr, q16, nullptr, M, Dm, Dm, 0, 0);
            gemm16<<<dim3(Dm / 128, M / 64), 256, 0, stream>>>(
                x16, wt16[l][1], P[l][3], nullptr, k16, nullptr, M, Dm, Dm, 0, 0);
            gemm16<<<dim3(Dm / 128, M / 64), 256, 0, stream>>>(
                x16, wt16[l][2], P[l][5], nullptr, nullptr, vt16, M, Dm, Dm, 0, Ls);
            attn_k<<<B * NHEAD * (L / 16), 128, 0, stream>>>(q16, k16, vt16, msk, a16, L);
            gemm16<<<dim3(Dm / 128, M / 64), 256, 0, stream>>>(
                a16, wt16[l][3], P[l][7], tmpf, nullptr, nullptr, M, Dm, Dm, 0, 0);
            add_ln_k<<<M, 256, 0, stream>>>(xf, tmpf, P[l][12], P[l][13], x16);
            gemm16<<<dim3(1024 / 128, M / 64), 256, 0, stream>>>(
                x16, wt16[l][4], P[l][9], nullptr, h16, nullptr, M, 1024, Dm, 1, 0);
            gemm16<<<dim3(Dm / 128, M / 64), 256, 0, stream>>>(
                h16, wt16[l][5], P[l][11], tmpf, nullptr, nullptr, M, Dm, 1024, 0, 0);
            add_ln_k<<<M, 256, 0, stream>>>(xf, tmpf, P[l][14], P[l][15], x16);
        }
        mix_norm_k<<<M, 256, 0, stream>>>(sembf, xf, msk, mixer, nout);
    };

    run_encoder(q_emb, q_mask, QL, 5, qn16);    // query side first (small)
    run_encoder(d_emb, d_mask, DL, 10, dn16);   // document side reuses scratch

    pos_mult_k<<<B, 256, 0, stream>>>(d_mask, pb, pba, multb);
    zero_f32<<<2, 256, 0, stream>>>(pk, B * NKER);
    score_k<<<MQ, 256, 0, stream>>>(qn16, dn16, q_mask, multb, pk);
    final_k<<<1, 32, 0, stream>>>(pk, dw, (float*)d_out);
}
